// XLSTMLayer_11785390260730
// MI455X (gfx1250) — compile-verified, matrix operand order checked by reading, not run on hardware
//
#include <hip/hip_runtime.h>
#include <stddef.h>

// ---------------------------------------------------------------------------
// xLSTM layer for MI455X (gfx1250): bf16 WMMA everywhere.
//   B=128, T=1024, D=128, U=256, M=64
// Phase 0: pack fp32 weights -> bf16 B-fragment-swizzled tiles (workspace)
// Phase 1: x projections -> bf16 stored directly in WMMA D-fragment order
// Phase 2: persistent per-batch-slice scan: 8 WGs x 16 batch rows, no grid sync
// ---------------------------------------------------------------------------

typedef __attribute__((ext_vector_type(16))) __bf16          v16bf;
typedef __attribute__((ext_vector_type(8)))  __bf16          v8bf;
typedef __attribute__((ext_vector_type(8)))  float           v8f;
typedef __attribute__((ext_vector_type(8)))  unsigned short  ushort8;

union Frag16 {            // one WMMA 16-bit operand fragment (8 VGPRs)
    v16bf   bf;
    __bf16  h[16];
    ushort8 us[2];
};
union Half8 {             // 8 bf16 = 16B (one b128 transaction)
    v8bf    bf;
    ushort8 us;
};

__device__ inline __bf16 f2bf(float x) { return (__bf16)x; }   // native cvt
__device__ inline unsigned short bfbits(__bf16 v) {
    union { __bf16 b; unsigned short s; } u; u.b = v; return u.s;
}
__device__ inline __bf16 bitsbf(unsigned short s) {
    union { __bf16 b; unsigned short s; } u; u.s = s; return u.b;
}

#if __has_builtin(__builtin_amdgcn_exp2f)
#define EXP2F(x) __builtin_amdgcn_exp2f(x)
#else
#define EXP2F(x) exp2f(x)
#endif
#if __has_builtin(__builtin_amdgcn_tanhf)
#define TANHF(x) __builtin_amdgcn_tanhf(x)
#else
#define TANHF(x) tanhf(x)
#endif
#if __has_builtin(__builtin_amdgcn_rcpf)
#define RCPF(x) __builtin_amdgcn_rcpf(x)
#else
#define RCPF(x) (1.0f / (x))
#endif

__device__ inline float sigmoidf(float x) {
    return RCPF(1.0f + EXP2F(-1.4426950408889634f * x));
}

static constexpr int Bsz = 128, Tn = 1024, Dn = 128, Un = 256, Mn = 64;
static constexpr int HPAD = 264;   // h row pitch (ushorts): 528B, bank-spread
static constexpr int MPAD = 72;    // mem row pitch (ushorts)

// ---------------------------------------------------------------------------
// Phase 0: pack a row-major [K][N] fp32 matrix into bf16 B-fragments.
// Tile (kt,nt): rows kt*32..+31, cols nt*16..+15.  Tile order: nt-major.
//   n = nt*16 + (l&15)
//   k = kt*32 + (l<16 ? (e<8? e : e+8) : (e<8? e+8 : e+16))
// ---------------------------------------------------------------------------
__global__ __launch_bounds__(256) void pack_b_kernel(
    const float* __restrict__ src, unsigned short* __restrict__ dst,
    int K, int N)
{
    int idx = blockIdx.x * blockDim.x + threadIdx.x;
    int KT = K >> 5;
    int total = KT * (N >> 4) * 512;
    if (idx >= total) return;
    int e    = idx & 15;
    int lane = (idx >> 4) & 31;
    int tile = idx >> 9;
    int kt = tile % KT, nt = tile / KT;
    int n = nt * 16 + (lane & 15);
    int koff = (lane < 16) ? ((e < 8) ? e : e + 8)
                           : ((e < 8) ? e + 8 : e + 16);
    int k = kt * 32 + koff;
    dst[idx] = bfbits(f2bf(src[(size_t)k * N + n]));
}

// ---------------------------------------------------------------------------
// Phase 1: x projections.  grid = (B/16)*T blocks, 256 thr (8 waves).
// Wave w: gate = w&3, column half = w>>2 (128 cols = 8 ntiles).
// Output: bf16 in WMMA D-fragment order:
//   xg[g][t][bslice(8)][ntile(16)][lane(32)][v(8)]
// so the scan can fetch one b128 per (gate, ntile).  Bias folded in.
// ---------------------------------------------------------------------------
__global__ __launch_bounds__(256) void xproj_kernel(
    const float* __restrict__ X, const unsigned short* __restrict__ Wp,
    const float* __restrict__ b_i, const float* __restrict__ b_f,
    const float* __restrict__ b_c, const float* __restrict__ b_o,
    unsigned short* __restrict__ xg)
{
    int blk = blockIdx.x;
    int t   = blk >> 3;
    int bsl = blk & 7;
    int b0  = bsl << 4;
    int tid = threadIdx.x, w = tid >> 5, lane = tid & 31;
    int g = w & 3, half = w >> 2;
    int lrow = lane & 15, lhi = lane >> 4;
    const float* bias = (g == 0) ? b_i : (g == 1) ? b_f : (g == 2) ? b_c : b_o;

    // A fragments (16 rows of X at fixed t), fp32 -> bf16 on the fly
    Frag16 aX[4];
    const float* xrow = X + ((size_t)(b0 + lrow) * Tn + t) * Dn;
#pragma unroll
    for (int kt = 0; kt < 4; ++kt) {
        int kb = kt * 32 + lhi * 8;
#pragma unroll
        for (int e = 0; e < 8; ++e) {
            aX[kt].h[e]     = f2bf(xrow[kb + e]);
            aX[kt].h[e + 8] = f2bf(xrow[kb + 16 + e]);
        }
    }

    const unsigned short* wg = Wp + (size_t)g * (4 * 16 * 512);
    // base of this (g, t, bslice): stride per ntile = 32*8 = 256 elems
    unsigned short* outp =
        xg + (((size_t)g * Tn + t) * 8 + bsl) * 4096 + (size_t)lane * 8;
#pragma unroll
    for (int nt = 0; nt < 8; ++nt) {
        int ntg = half * 8 + nt;
        v8f acc = {};
        const unsigned short* bp = wg + ((size_t)ntg * 4) * 512 + lane * 16;
#pragma unroll
        for (int kt = 0; kt < 4; ++kt) {
            Frag16 bfr = *(const Frag16*)(bp + kt * 512);
            acc = __builtin_amdgcn_wmma_f32_16x16x32_bf16(
                false, aX[kt].bf, false, bfr.bf, (short)0, acc, false, false);
        }
        float bv = bias[ntg * 16 + lrow];
        Half8 st;
#pragma unroll
        for (int v = 0; v < 8; ++v) st.bf[v] = f2bf(acc[v] + bv);
        *(ushort8*)(outp + (size_t)ntg * 256) = st.us;
    }
}

// ---------------------------------------------------------------------------
// Phase 2: persistent scan.  8 blocks (one per 16-row batch slice), 8 waves.
// Wave w owns output columns [w*32, w*32+32): 2 ntiles, c-state in registers.
// h / mem live in double-buffered bf16 LDS; weights stream from L2-hot global.
// ---------------------------------------------------------------------------
__global__ __launch_bounds__(256, 1) void scan_kernel(
    const unsigned short* __restrict__ Up,
    const unsigned short* __restrict__ Wmemp,
    const unsigned short* __restrict__ Umemp,
    const unsigned short* __restrict__ xg,
    float* __restrict__ out)
{
    __shared__ unsigned short h_bf[2][16][HPAD];
    __shared__ unsigned short mem_bf[2][16][MPAD];
    __shared__ float          mem_f[16][Mn];

    int tid = threadIdx.x, w = tid >> 5, lane = tid & 31;
    int lrow = lane & 15, lhi = lane >> 4;
    int r0 = blockIdx.x << 4;
    int n0 = w * 32;

    for (int i = tid; i < 2 * 16 * HPAD; i += 256) (&h_bf[0][0][0])[i] = 0;
    for (int i = tid; i < 2 * 16 * MPAD; i += 256) (&mem_bf[0][0][0])[i] = 0;
    for (int i = tid; i < 16 * Mn; i += 256) (&mem_f[0][0])[i] = 0.0f;
    __syncthreads();

    v8f cst[2];
    cst[0] = {}; cst[1] = {};

    // ---- loop-invariant pointers (weights, x-fragments, output) -----------
    const unsigned short* bpU[4][2];
#pragma unroll
    for (int g = 0; g < 4; ++g)
#pragma unroll
        for (int nt = 0; nt < 2; ++nt)
            bpU[g][nt] = Up + (size_t)g * (8 * 16 * 512) +
                         ((size_t)(w * 2 + nt) * 8) * 512 + lane * 16;
    const unsigned short* bpWm[2];
#pragma unroll
    for (int nt = 0; nt < 2; ++nt)
        bpWm[nt] = Wmemp + ((size_t)(w * 2 + nt) * 2) * 512 + lane * 16;
    const unsigned short* bpUm =
        Umemp + ((size_t)(w & 3) * 8) * 512 + lane * 16;

    const unsigned short* xp[4][2];   // advance by 32768 elems per step
#pragma unroll
    for (int g = 0; g < 4; ++g)
#pragma unroll
        for (int nt = 0; nt < 2; ++nt)
            xp[g][nt] = xg + (((size_t)g * Tn) * 8 + blockIdx.x) * 4096 +
                        (size_t)(w * 2 + nt) * 256 + (size_t)lane * 8;

    float* op[2];                     // advance by Un floats per step
#pragma unroll
    for (int nt = 0; nt < 2; ++nt)
        op[nt] = out + ((size_t)(r0 + lhi * 8) * Tn) * Un + n0 + nt * 16 + lrow;

    for (int t = 0; t < Tn; ++t) {
        int p = t & 1;

        // ---- A fragments: h_prev (8 ktiles) and mem_prev (2 ktiles) -------
        Frag16 aH[8];
#pragma unroll
        for (int kt = 0; kt < 8; ++kt) {
            int kb = kt * 32 + lhi * 8;
            aH[kt].us[0] = *(const ushort8*)&h_bf[p][lrow][kb];
            aH[kt].us[1] = *(const ushort8*)&h_bf[p][lrow][kb + 16];
        }
        Frag16 aM[2];
#pragma unroll
        for (int kt = 0; kt < 2; ++kt) {
            int kb = kt * 32 + lhi * 8;
            aM[kt].us[0] = *(const ushort8*)&mem_bf[p][lrow][kb];
            aM[kt].us[1] = *(const ushort8*)&mem_bf[p][lrow][kb + 16];
        }

        // ---- x fragments for this step (one b128 per gate/ntile) ----------
        Half8 xv[4][2];
#pragma unroll
        for (int g = 0; g < 4; ++g)
#pragma unroll
            for (int nt = 0; nt < 2; ++nt) {
                xv[g][nt].us = *(const ushort8*)xp[g][nt];
                xp[g][nt] += 32768;
            }

        // ---- 4 recurrent GEMMs: acc[g][nt] = h_prev @ U_g -----------------
        v8f acc[4][2];
#pragma unroll
        for (int g = 0; g < 4; ++g)
#pragma unroll
            for (int nt = 0; nt < 2; ++nt) {
                v8f a = {};
#pragma unroll
                for (int kt = 0; kt < 8; ++kt) {
                    Frag16 bfr = *(const Frag16*)(bpU[g][nt] + kt * 512);
                    a = __builtin_amdgcn_wmma_f32_16x16x32_bf16(
                        false, aH[kt].bf, false, bfr.bf, (short)0, a, false, false);
                }
                acc[g][nt] = a;
            }
        // ---- mem_read = mem_prev @ W_mem adds into gate i -----------------
#pragma unroll
        for (int nt = 0; nt < 2; ++nt)
#pragma unroll
            for (int kt = 0; kt < 2; ++kt) {
                Frag16 bfr = *(const Frag16*)(bpWm[nt] + kt * 512);
                acc[0][nt] = __builtin_amdgcn_wmma_f32_16x16x32_bf16(
                    false, aM[kt].bf, false, bfr.bf, (short)0, acc[0][nt],
                    false, false);
            }

        // prefetch next step's x fragments (global_prefetch_b8)
        if (t + 1 < Tn) {
#pragma unroll
            for (int g = 0; g < 4; ++g)
                __builtin_prefetch(xp[g][0], 0, 1);
        }

        // ---- gates / cell / hidden ----------------------------------------
#pragma unroll
        for (int nt = 0; nt < 2; ++nt) {
            int col = n0 + nt * 16 + lrow;
#pragma unroll
            for (int v = 0; v < 8; ++v) {
                int m = v + lhi * 8;
                float iv = sigmoidf(acc[0][nt][v] + (float)xv[0][nt].bf[v]);
                float fv = sigmoidf(acc[1][nt][v] + (float)xv[1][nt].bf[v]);
                float ct = TANHF(acc[2][nt][v] + (float)xv[2][nt].bf[v]);
                float ov = sigmoidf(acc[3][nt][v] + (float)xv[3][nt].bf[v]);
                float cn = fv * cst[nt][v] + iv * ct;
                cst[nt][v] = cn;
                float hn = ov * TANHF(cn);
                op[nt][(size_t)v * Tn * Un] = hn;
                h_bf[p ^ 1][m][col] = bfbits(f2bf(hn));
            }
            op[nt] += Un;
        }
        __syncthreads();   // h_new complete in LDS

        // ---- mem_new = mem_prev + 0.1*(h_new @ U_mem): waves 0..3 ---------
        if (w < 4) {
            Frag16 aH2[8];
#pragma unroll
            for (int kt = 0; kt < 8; ++kt) {
                int kb = kt * 32 + lhi * 8;
                aH2[kt].us[0] = *(const ushort8*)&h_bf[p ^ 1][lrow][kb];
                aH2[kt].us[1] = *(const ushort8*)&h_bf[p ^ 1][lrow][kb + 16];
            }
            v8f a = {};
#pragma unroll
            for (int kt = 0; kt < 8; ++kt) {
                Frag16 bfr = *(const Frag16*)(bpUm + kt * 512);
                a = __builtin_amdgcn_wmma_f32_16x16x32_bf16(
                    false, aH2[kt].bf, false, bfr.bf, (short)0, a, false, false);
            }
#pragma unroll
            for (int v = 0; v < 8; ++v) {
                int m = v + lhi * 8;
                int col = w * 16 + lrow;
                float nm = mem_f[m][col] + 0.1f * a[v];
                mem_f[m][col] = nm;
                mem_bf[p ^ 1][m][col] = bfbits(f2bf(nm));
            }
        }
        __syncthreads();   // mem_new complete for next step
    }
}

// ---------------------------------------------------------------------------
// Host-side launcher (graph-capture safe: launches only, all on `stream`)
// ---------------------------------------------------------------------------
extern "C" void kernel_launch(void* const* d_in, const int* in_sizes, int n_in,
                              void* d_out, int out_size, void* d_ws, size_t ws_size,
                              hipStream_t stream)
{
    (void)in_sizes; (void)n_in; (void)out_size; (void)ws_size;

    const float* X    = (const float*)d_in[0];
    const float* Wg[4]= {(const float*)d_in[1], (const float*)d_in[2],
                         (const float*)d_in[3], (const float*)d_in[4]};
    const float* Ug[4]= {(const float*)d_in[5], (const float*)d_in[6],
                         (const float*)d_in[7], (const float*)d_in[8]};
    const float* Wmem = (const float*)d_in[9];
    const float* Umem = (const float*)d_in[10];
    const float* b_i  = (const float*)d_in[11];
    const float* b_f  = (const float*)d_in[12];
    const float* b_c  = (const float*)d_in[13];
    const float* b_o  = (const float*)d_in[14];
    float* out = (float*)d_out;

    // Workspace layout (bytes)
    unsigned char* ws = (unsigned char*)d_ws;
    const size_t W_ELEMS  = 4 * 16 * 512;   // per gate, K=128 (KT=4, NT=16)
    const size_t U_ELEMS  = 8 * 16 * 512;   // per gate, K=256 (KT=8, NT=16)
    const size_t WM_ELEMS = 2 * 16 * 512;   // K=64 -> U=256
    const size_t UM_ELEMS = 8 * 4 * 512;    // K=256 -> M=64
    unsigned short* Wp    = (unsigned short*)(ws + 0);
    unsigned short* Upk   = (unsigned short*)(ws + 4 * W_ELEMS * 2);
    unsigned short* Wmemp = (unsigned short*)(ws + 4 * W_ELEMS * 2 + 4 * U_ELEMS * 2);
    unsigned short* Umemp = (unsigned short*)(ws + 4 * W_ELEMS * 2 + 4 * U_ELEMS * 2
                                                 + WM_ELEMS * 2);
    unsigned short* xg    = (unsigned short*)(ws + (1u << 20));  // 1MB-aligned

    // Phase 0: pack weights to bf16 B-fragment layout
    for (int g = 0; g < 4; ++g)
        pack_b_kernel<<<(int)(W_ELEMS / 256), 256, 0, stream>>>(
            Wg[g], Wp + (size_t)g * W_ELEMS, Dn, Un);
    for (int g = 0; g < 4; ++g)
        pack_b_kernel<<<(int)(U_ELEMS / 256), 256, 0, stream>>>(
            Ug[g], Upk + (size_t)g * U_ELEMS, Un, Un);
    pack_b_kernel<<<(int)(WM_ELEMS / 256), 256, 0, stream>>>(Wmem, Wmemp, Mn, Un);
    pack_b_kernel<<<(int)(UM_ELEMS / 256), 256, 0, stream>>>(Umem, Umemp, Un, Mn);

    // Phase 1: x projections (bias folded, fragment-order output)
    xproj_kernel<<<(Bsz / 16) * Tn, 256, 0, stream>>>(X, Wp, b_i, b_f, b_c, b_o, xg);

    // Phase 2: recurrent scan, one workgroup per 16-row batch slice
    scan_kernel<<<Bsz / 16, 256, 0, stream>>>(Upk, Wmemp, Umemp, xg, out);
}
